// BMHA_91276644974891
// MI455X (gfx1250) — compile-verified, matrix operand order checked by reading, not run on hardware
//
#include <hip/hip_runtime.h>
#include <hip/hip_bf16.h>

typedef __attribute__((ext_vector_type(16))) _Float16 v16h;
typedef __attribute__((ext_vector_type(8)))  _Float16 v8h;
typedef __attribute__((ext_vector_type(8)))  float    v8f;

#define WMMA_F32_F16(a, b, c) \
    __builtin_amdgcn_wmma_f32_16x16x32_f16(false, (a), false, (b), (short)0, (c), false, false)

// ---------------------------------------------------------------------------
// Fragment loader for V_WMMA_F32_16X16X32_F16 A/B operands.
// 16-bit 16x32 fragment layout (ISA 7.12.2): lanes 0-15 hold row r=lane with
// K in {0..7, 16..23}; lanes 16-31 hold row r=lane-16 with K in {8..15, 24..31}.
// Both A (rows of X) and B (columns == rows of W for the X*W^T pattern) load
// identically: two 16-byte vector loads per lane.
// ---------------------------------------------------------------------------
__device__ __forceinline__ v16h load_frag_g(const _Float16* __restrict__ base, int ld) {
    int lane = threadIdx.x & 31;
    int r    = lane & 15;
    int kb   = (lane >> 4) << 3;   // 0 or 8
    const _Float16* p = base + (size_t)r * ld + kb;
    v8h lo = *(const v8h*)(p);       // K = kb .. kb+7
    v8h hi = *(const v8h*)(p + 16);  // K = kb+16 .. kb+23
    return __builtin_shufflevector(lo, hi, 0,1,2,3,4,5,6,7,8,9,10,11,12,13,14,15);
}

// ---------------------------------------------------------------------------
// Convert / binarize: f32 -> f16, optionally sign-binarize (+1/-1).
// ---------------------------------------------------------------------------
__global__ void cvt_kernel(const float* __restrict__ src, _Float16* __restrict__ dst,
                           int n, int binarize) {
    int i = blockIdx.x * blockDim.x + threadIdx.x;
    if (i < n) {
        float v = src[i];
        if (binarize) v = (v >= 0.0f) ? 1.0f : -1.0f;
        dst[i] = (_Float16)v;
    }
}

// ---------------------------------------------------------------------------
// Binary projection GEMM: C[M=4096, N=1024] = X[4096,1024] * W[1024,1024]^T
// (W pre-binarized f16). Per wave: 32(M) x 64(N) tile, K-loop of 32,
// 8 WMMAs per K-step. Block = 4 waves covering 128 rows.
// Store modes: 0 = Q/K head-split [B,H,S,hd] f16
//              2 = V transposed  [B,H,hd,S] f16
//              3 = plain f32 [M,N] (final output projection)
// ---------------------------------------------------------------------------
__global__ void gemm_bin_kernel(const _Float16* __restrict__ X,
                                const _Float16* __restrict__ W,
                                void* __restrict__ outp, int mode) {
    const int K = 1024;
    int lane = threadIdx.x & 31;
    int wave = threadIdx.y;
    int m0 = blockIdx.x * 128 + wave * 32;
    int n0 = blockIdx.y * 64;

    v8f acc[2][4];
    #pragma unroll
    for (int i = 0; i < 2; ++i)
        #pragma unroll
        for (int t = 0; t < 4; ++t)
            acc[i][t] = (v8f){};

    for (int kk = 0; kk < K; kk += 32) {
        v16h a0 = load_frag_g(X + (size_t)m0 * K + kk, K);
        v16h a1 = load_frag_g(X + (size_t)(m0 + 16) * K + kk, K);
        v16h b0 = load_frag_g(W + (size_t)(n0 +  0) * K + kk, K);
        v16h b1 = load_frag_g(W + (size_t)(n0 + 16) * K + kk, K);
        v16h b2 = load_frag_g(W + (size_t)(n0 + 32) * K + kk, K);
        v16h b3 = load_frag_g(W + (size_t)(n0 + 48) * K + kk, K);
        acc[0][0] = WMMA_F32_F16(a0, b0, acc[0][0]);
        acc[0][1] = WMMA_F32_F16(a0, b1, acc[0][1]);
        acc[0][2] = WMMA_F32_F16(a0, b2, acc[0][2]);
        acc[0][3] = WMMA_F32_F16(a0, b3, acc[0][3]);
        acc[1][0] = WMMA_F32_F16(a1, b0, acc[1][0]);
        acc[1][1] = WMMA_F32_F16(a1, b1, acc[1][1]);
        acc[1][2] = WMMA_F32_F16(a1, b2, acc[1][2]);
        acc[1][3] = WMMA_F32_F16(a1, b3, acc[1][3]);
    }

    // C layout: acc vgpr j -> M = j (lanes 0-15) or j+8 (lanes 16-31); N = lane&15
    int hi8 = (lane >> 4) << 3;
    int nl  = lane & 15;
    #pragma unroll
    for (int i = 0; i < 2; ++i) {
        #pragma unroll
        for (int t = 0; t < 4; ++t) {
            #pragma unroll
            for (int j = 0; j < 8; ++j) {
                int m = m0 + i * 16 + j + hi8;     // global row (b*2048+s)
                int n = n0 + t * 16 + nl;          // global col (h*64+d)
                float v = acc[i][t][j];
                if (mode == 3) {
                    ((float*)outp)[(size_t)m * 1024 + n] = v;
                } else {
                    int b = m >> 11, s = m & 2047;
                    int h = n >> 6,  d = n & 63;
                    _Float16* o16 = (_Float16*)outp;
                    if (mode == 2) {
                        // V^T: [B,H,hd,S]
                        o16[(((size_t)(b * 16 + h) * 64 + d) * 2048) + s] = (_Float16)v;
                    } else {
                        // Q/K: [B,H,S,hd]
                        o16[(((size_t)(b * 16 + h) * 2048 + s) * 64) + d] = (_Float16)v;
                    }
                }
            }
        }
    }
}

// ---------------------------------------------------------------------------
// Fused attention, one wave per (b, h, 16-query strip).
// Pass 1: online softmax stats (row max, row sumexp) from QK^T WMMAs.
// Pass 2: recompute scores, write probs f32 exactly once, stage f16 probs in
//         1KB LDS tile (lane transpose acc->A-frag), accumulate PV WMMAs.
// Q,K: [B,H,S,64] f16.  V^T: [B,H,64,S] f16.  attn out: [B,S,1024] f16.
// ---------------------------------------------------------------------------
__global__ void attn_kernel(const _Float16* __restrict__ Q,
                            const _Float16* __restrict__ Km,
                            const _Float16* __restrict__ Vt,
                            const int* __restrict__ mask,
                            float* __restrict__ probs,
                            _Float16* __restrict__ attn) {
    const int S = 2048, HD = 64;
    int idx = blockIdx.x;
    int qt = idx & 127;
    int h  = (idx >> 7) & 15;
    int b  = idx >> 11;
    int q0 = qt * 16;
    int lane = threadIdx.x & 31;
    int hi8 = (lane >> 4) << 3;
    int nl  = lane & 15;

    const _Float16* Qh = Q  + ((size_t)(b * 16 + h) * S) * HD;
    const _Float16* Kh = Km + ((size_t)(b * 16 + h) * S) * HD;
    const _Float16* Vh = Vt + ((size_t)(b * 16 + h) * HD) * S;
    const int* mk = mask + b * S;

    // Q fragments for this strip (hd = 64 -> two 32-wide K steps), reused all passes
    v16h qa0 = load_frag_g(Qh + (size_t)q0 * HD +  0, HD);
    v16h qa1 = load_frag_g(Qh + (size_t)q0 * HD + 32, HD);

    const float scale = 0.125f;  // 1/sqrt(64)

    // ---- Pass 1: online row max / sumexp over this lane's key columns ----
    float mrow[8], lrow[8];
    #pragma unroll
    for (int j = 0; j < 8; ++j) { mrow[j] = -3.0e38f; lrow[j] = 0.0f; }

    for (int kt = 0; kt < S / 16; ++kt) {
        const _Float16* kbase = Kh + (size_t)(kt * 16) * HD;
        __builtin_prefetch(Kh + (size_t)((kt + 1) & (S/16 - 1)) * 16 * HD + (size_t)nl * HD, 0, 1);
        v16h kb0 = load_frag_g(kbase +  0, HD);
        v16h kb1 = load_frag_g(kbase + 32, HD);
        v8f c = (v8f){};
        c = WMMA_F32_F16(qa0, kb0, c);
        c = WMMA_F32_F16(qa1, kb1, c);
        int mv = mk[kt * 16 + nl];
        #pragma unroll
        for (int j = 0; j < 8; ++j) {
            float s = mv ? c[j] * scale : -1.0e9f;
            float mn = fmaxf(mrow[j], s);
            lrow[j] = lrow[j] * __expf(mrow[j] - mn) + __expf(s - mn);
            mrow[j] = mn;
        }
    }
    // Merge across the 16 lanes of each half-wave (halves hold different rows)
    #pragma unroll
    for (int off = 8; off >= 1; off >>= 1) {
        #pragma unroll
        for (int j = 0; j < 8; ++j) {
            float mo = __shfl_xor(mrow[j], off, 32);
            float lo = __shfl_xor(lrow[j], off, 32);
            float mn = fmaxf(mrow[j], mo);
            lrow[j] = lrow[j] * __expf(mrow[j] - mn) + lo * __expf(mo - mn);
            mrow[j] = mn;
        }
    }
    float invl[8];
    #pragma unroll
    for (int j = 0; j < 8; ++j) invl[j] = 1.0f / lrow[j];

    // ---- Pass 2: recompute scores, emit probs, accumulate PV ----
    __shared__ __align__(16) _Float16 pstage[16][40];  // 16x32 tile, padded rows
    v8f acc[4];
    #pragma unroll
    for (int t = 0; t < 4; ++t) acc[t] = (v8f){};

    float* prow = probs + (((size_t)(b * 16 + h) * S) + q0) * S;

    for (int kt2 = 0; kt2 < S / 32; ++kt2) {
        #pragma unroll
        for (int half = 0; half < 2; ++half) {
            int kt = kt2 * 2 + half;
            const _Float16* kbase = Kh + (size_t)(kt * 16) * HD;
            v16h kb0 = load_frag_g(kbase +  0, HD);
            v16h kb1 = load_frag_g(kbase + 32, HD);
            v8f c = (v8f){};
            c = WMMA_F32_F16(qa0, kb0, c);
            c = WMMA_F32_F16(qa1, kb1, c);
            int mv = mk[kt * 16 + nl];
            #pragma unroll
            for (int j = 0; j < 8; ++j) {
                float s = mv ? c[j] * scale : -1.0e9f;
                float p = __expf(s - mrow[j]) * invl[j];
                int m = j + hi8;
                prow[(size_t)m * S + kt * 16 + nl] = p;        // probs written once
                pstage[m][half * 16 + nl] = (_Float16)p;       // stage for PV A-frag
            }
        }
        __syncthreads();
        // A-frag (probs 16x32) from LDS, standard split-K layout
        v16h pa;
        {
            const _Float16* p = &pstage[nl][hi8];
            v8h lo = *(const v8h*)(p);
            v8h hi = *(const v8h*)(p + 16);
            pa = __builtin_shufflevector(lo, hi, 0,1,2,3,4,5,6,7,8,9,10,11,12,13,14,15);
        }
        // B-frags from V^T (contiguous along S), one per 16-wide d tile
        #pragma unroll
        for (int t = 0; t < 4; ++t) {
            v16h vb = load_frag_g(Vh + (size_t)(t * 16) * S + kt2 * 32, S);
            acc[t] = WMMA_F32_F16(pa, vb, acc[t]);
        }
        __syncthreads();
    }

    // attn out: [B, S, H*64] f16 (natural GEMM layout for out-projection)
    #pragma unroll
    for (int t = 0; t < 4; ++t) {
        #pragma unroll
        for (int j = 0; j < 8; ++j) {
            int m = j + hi8;
            int row = q0 + m;
            int col = h * 64 + t * 16 + nl;
            attn[((size_t)(b * S + row)) * 1024 + col] = (_Float16)acc[t][j];
        }
    }
}

// ---------------------------------------------------------------------------
extern "C" void kernel_launch(void* const* d_in, const int* in_sizes, int n_in,
                              void* d_out, int out_size, void* d_ws, size_t ws_size,
                              hipStream_t stream) {
    const float* hidden = (const float*)d_in[0];
    const int*   mask   = (const int*)  d_in[1];
    const float* wq     = (const float*)d_in[2];
    const float* wk     = (const float*)d_in[3];
    const float* wv     = (const float*)d_in[4];
    const float* wo     = (const float*)d_in[5];

    const int BS = 2 * 2048;          // 4096 rows
    const int D  = 1024;
    const int NX = BS * D;            // 4,194,304
    const int NW = D * D;             // 1,048,576

    _Float16* Xh   = (_Float16*)d_ws;
    _Float16* Wqb  = Xh   + NX;
    _Float16* Wkb  = Wqb  + NW;
    _Float16* Wvb  = Wkb  + NW;
    _Float16* Wob  = Wvb  + NW;
    _Float16* Qb   = Wob  + NW;       // [B,H,S,hd]
    _Float16* Kb   = Qb   + NX;       // [B,H,S,hd]
    _Float16* Vtb  = Kb   + NX;       // [B,H,hd,S]
    _Float16* attn = Vtb  + NX;       // [B,S,D]

    float* outp  = (float*)d_out;     // [B,S,D]
    float* probs = outp + NX;         // [B,H,S,S]

    cvt_kernel<<<(NX + 255) / 256, 256, 0, stream>>>(hidden, Xh, NX, 0);
    cvt_kernel<<<(NW + 255) / 256, 256, 0, stream>>>(wq, Wqb, NW, 1);
    cvt_kernel<<<(NW + 255) / 256, 256, 0, stream>>>(wk, Wkb, NW, 1);
    cvt_kernel<<<(NW + 255) / 256, 256, 0, stream>>>(wv, Wvb, NW, 1);
    cvt_kernel<<<(NW + 255) / 256, 256, 0, stream>>>(wo, Wob, NW, 1);

    dim3 gblk(32, 4);
    dim3 ggrd(BS / 128, D / 64);      // 32 x 16
    gemm_bin_kernel<<<ggrd, gblk, 0, stream>>>(Xh, Wqb, (void*)Qb, 0);
    gemm_bin_kernel<<<ggrd, gblk, 0, stream>>>(Xh, Wkb, (void*)Kb, 0);
    gemm_bin_kernel<<<ggrd, gblk, 0, stream>>>(Xh, Wvb, (void*)Vtb, 2);

    attn_kernel<<<2 * 16 * (2048 / 16), 32, 0, stream>>>(Qb, Kb, Vtb, mask, probs, attn);

    gemm_bin_kernel<<<ggrd, gblk, 0, stream>>>(attn, Wob, (void*)outp, 3);
}
